// Attention_76974403879611
// MI455X (gfx1250) — compile-verified, matrix operand order checked by reading, not run on hardware
//
#include <hip/hip_runtime.h>

typedef __attribute__((ext_vector_type(16))) __bf16 v16bf;
typedef __attribute__((ext_vector_type(8)))  float  v8f;

#define SLEN  2048
#define DHEAD 128

template <bool B> struct BoolC { static constexpr bool value = B; };

union FragA {
    v16bf v;
    unsigned int u[8];
};

// Packed f32x2 -> bf16x2 conversion (single VALU op).
__device__ __forceinline__ unsigned int cvt_pk_bf16(float a, float b) {
    unsigned int r;
    asm("v_cvt_pk_bf16_f32 %0, %1, %2" : "=v"(r) : "v"(a), "v"(b));
    return r;
}

// XOR-swap lane shuffles via ds_swizzle immediate patterns (group-of-32 mode:
// offset[14:10]=xor, [4:0]=and). No per-shuffle address math.
#define SWIZ(x, imm) __int_as_float(__builtin_amdgcn_ds_swizzle(__float_as_int(x), (imm)))

__device__ __forceinline__ float red_max16(float x) {   // max across 16-lane halves
    x = fmaxf(x, SWIZ(x, 0x201f));  // SWAPX8
    x = fmaxf(x, SWIZ(x, 0x101f));  // SWAPX4
    x = fmaxf(x, SWIZ(x, 0x081f));  // SWAPX2
    x = fmaxf(x, SWIZ(x, 0x041f));  // SWAPX1
    return x;
}
__device__ __forceinline__ float red_sum16(float x) {   // sum across 16-lane halves
    x += SWIZ(x, 0x201f);
    x += SWIZ(x, 0x101f);
    x += SWIZ(x, 0x081f);
    x += SWIZ(x, 0x041f);
    return x;
}

// Load a 16-bit WMMA A/B fragment: two contiguous 8-element (16B) runs per lane.
__device__ __forceinline__ v16bf lds_frag(const unsigned short* base, int c1, int c2) {
    FragA f;
    uint4 a = *(const uint4*)(base + c1);
    uint4 b = *(const uint4*)(base + c2);
    f.u[0] = a.x; f.u[1] = a.y; f.u[2] = a.z; f.u[3] = a.w;
    f.u[4] = b.x; f.u[5] = b.y; f.u[6] = b.z; f.u[7] = b.w;
    return f.v;
}

// Build a Q A-fragment from global fp32, folding in scale = log2(e)/sqrt(d).
__device__ __forceinline__ v16bf q_frag(const float* g, int c1, int c2, float s) {
    FragA f;
    float4 a0 = *(const float4*)(g + c1);
    float4 a1 = *(const float4*)(g + c1 + 4);
    float4 b0 = *(const float4*)(g + c2);
    float4 b1 = *(const float4*)(g + c2 + 4);
    f.u[0] = cvt_pk_bf16(a0.x * s, a0.y * s); f.u[1] = cvt_pk_bf16(a0.z * s, a0.w * s);
    f.u[2] = cvt_pk_bf16(a1.x * s, a1.y * s); f.u[3] = cvt_pk_bf16(a1.z * s, a1.w * s);
    f.u[4] = cvt_pk_bf16(b0.x * s, b0.y * s); f.u[5] = cvt_pk_bf16(b0.z * s, b0.w * s);
    f.u[6] = cvt_pk_bf16(b1.x * s, b1.y * s); f.u[7] = cvt_pk_bf16(b1.z * s, b1.w * s);
    return f.v;
}

__global__ __launch_bounds__(256)
void fa_fwd_kernel(const float* __restrict__ Kg, const float* __restrict__ Qg,
                   const float* __restrict__ Vg, float* __restrict__ Og) {
    __shared__ alignas(16) unsigned short Kl[32 * 128];    // K tile, row-major bf16
    __shared__ alignas(16) unsigned short Vt[128 * 32];    // V tile, transposed bf16
    __shared__ alignas(16) unsigned short Pl[8 * 16 * 32]; // per-wave P tile bf16

    const int tid  = threadIdx.x;
    const int w    = tid >> 5;
    const int lane = tid & 31;
    const int half = lane >> 4;   // 0: lanes 0-15, 1: lanes 16-31
    const int lq   = lane & 15;
    const int q0   = blockIdx.x * 128;
    const int bh   = blockIdx.y;
    const int qrow = q0 + 16 * w;          // this wave's first query row

    const size_t base = (size_t)bh * SLEN * DHEAD;
    const float* Qb = Qg + base;
    const float* Kb = Kg + base;
    const float* Vb = Vg + base;
    float*       Ob = Og + base;

    // Staging addresses for this thread: 4 float4 groups cover the 32x128 tile.
    int srow[4], scol[4];
    #pragma unroll
    for (int j = 0; j < 4; ++j) {
        int flat = (tid + 256 * j) * 4;
        srow[j] = flat >> 7;
        scol[j] = flat & 127;
    }

    // Preload Q fragments; scale folds 1/sqrt(d) AND log2(e) (exp2 domain).
    v16bf qf[4];
    {
        const float scale = 0.08838834764831845f * 1.4426950408889634f;
        const float* qp = Qb + (size_t)(qrow + lq) * DHEAD;
        #pragma unroll
        for (int c = 0; c < 4; ++c)
            qf[c] = q_frag(qp, 32 * c + half * 8, 32 * c + 16 + half * 8, scale);
    }

    v8f zf = {};
    v8f acc[8];
    #pragma unroll
    for (int i = 0; i < 8; ++i) acc[i] = zf;
    float rowmax[8], rowsum[8];   // rowsum is a PER-LANE partial; reduced in epilogue
    #pragma unroll
    for (int i = 0; i < 8; ++i) { rowmax[i] = -1e30f; rowsum[i] = 0.0f; }

    const int my_limit = qrow + 15;            // last key visible to this wave
    const int ntiles   = (q0 + 128) / 32;
    unsigned short* Pw = Pl + w * 16 * 32;

    // Software pipeline: registers hold the next tile's raw fp32 K/V data.
    float4 kreg[4], vreg[4];
    #pragma unroll
    for (int j = 0; j < 4; ++j) {
        kreg[j] = *(const float4*)(Kb + (size_t)srow[j] * DHEAD + scol[j]);
        vreg[j] = *(const float4*)(Vb + (size_t)srow[j] * DHEAD + scol[j]);
    }

    for (int t = 0; t < ntiles; ++t) {
        const int k0 = 32 * t;
        __syncthreads();  // previous tile fully consumed

        // ---- convert prefetched registers -> LDS (K row-major, V transposed) ----
        #pragma unroll
        for (int j = 0; j < 4; ++j) {
            int r = srow[j], c = scol[j];
            uint2 kk;
            kk.x = cvt_pk_bf16(kreg[j].x, kreg[j].y);
            kk.y = cvt_pk_bf16(kreg[j].z, kreg[j].w);
            *(uint2*)(Kl + r * 128 + c) = kk;
            unsigned int v01 = cvt_pk_bf16(vreg[j].x, vreg[j].y);
            unsigned int v23 = cvt_pk_bf16(vreg[j].z, vreg[j].w);
            Vt[(c + 0) * 32 + r] = (unsigned short)v01;
            Vt[(c + 1) * 32 + r] = (unsigned short)(v01 >> 16);
            Vt[(c + 2) * 32 + r] = (unsigned short)v23;
            Vt[(c + 3) * 32 + r] = (unsigned short)(v23 >> 16);
        }
        __syncthreads();

        // ---- issue next tile's global loads now; wait happens next iteration ----
        if (t + 1 < ntiles) {
            const int kn = k0 + 32;
            #pragma unroll
            for (int j = 0; j < 4; ++j) {
                kreg[j] = *(const float4*)(Kb + (size_t)(kn + srow[j]) * DHEAD + scol[j]);
                vreg[j] = *(const float4*)(Vb + (size_t)(kn + srow[j]) * DHEAD + scol[j]);
            }
        }

        if (k0 > my_limit) continue;   // wave-uniform causal skip (barriers stay matched)

        // ---- logits: even/odd key split so a row's (s0,s1) pair is adjacent in P ----
        // s0 columns n <-> keys 2n ; s1 columns n <-> keys 2n+1.
        v8f s0 = zf, s1 = zf;
        #pragma unroll
        for (int c = 0; c < 4; ++c) {
            v16bf kb0 = lds_frag(Kl + (2 * lq)     * 128, 32 * c + half * 8, 32 * c + 16 + half * 8);
            v16bf kb1 = lds_frag(Kl + (2 * lq + 1) * 128, 32 * c + half * 8, 32 * c + 16 + half * 8);
            s0 = __builtin_amdgcn_wmma_f32_16x16x32_bf16(false, qf[c], false, kb0, (short)0, s0, false, false);
            s1 = __builtin_amdgcn_wmma_f32_16x16x32_bf16(false, qf[c], false, kb1, (short)0, s1, false, false);
        }

        // ---- streaming softmax in exp2 domain; compile-time mask variant ----
        float p0[8], p1[8];
        auto softmax_tile = [&](auto mc) {
            constexpr bool MASKED = decltype(mc)::value;
            const int mrow = qrow + half * 8;
            #pragma unroll
            for (int v = 0; v < 8; ++v) {
                float a = s0[v];
                float b = s1[v];
                if (MASKED) {
                    int m = mrow + v;
                    a = (k0 + 2 * lq     > m) ? -1e30f : a;
                    b = (k0 + 2 * lq + 1 > m) ? -1e30f : b;
                }
                float tm   = red_max16(fmaxf(a, b));
                float mnew = fmaxf(rowmax[v], tm);
                float cf   = __builtin_amdgcn_exp2f(rowmax[v] - mnew);
                rowmax[v]  = mnew;
                float e0 = __builtin_amdgcn_exp2f(a - mnew);
                float e1 = __builtin_amdgcn_exp2f(b - mnew);
                rowsum[v] = rowsum[v] * cf + (e0 + e1);   // per-lane partial sum
                p0[v] = e0; p1[v] = e1;
                #pragma unroll
                for (int nb = 0; nb < 8; ++nb) acc[nb][v] *= cf;
            }
        };
        if (k0 + 31 > qrow) softmax_tile(BoolC<true>{});   // diagonal tile(s) only
        else                softmax_tile(BoolC<false>{});  // common path: no mask VALU

        // ---- P (C-layout) -> LDS: adjacent columns 2lq,2lq+1 => one b32 store ----
        #pragma unroll
        for (int v = 0; v < 8; ++v) {
            unsigned int pk = cvt_pk_bf16(p0[v], p1[v]);
            *(unsigned int*)(Pw + (v + half * 8) * 32 + 2 * lq) = pk;
        }
        asm volatile("s_wait_dscnt 0" ::: "memory");

        // P column n == key n, so the A-fragment read and V layout are unchanged.
        v16bf pf = lds_frag(Pw + lq * 32, half * 8, 16 + half * 8);
        #pragma unroll
        for (int nb = 0; nb < 8; ++nb) {
            v16bf vf = lds_frag(Vt + (16 * nb + lq) * 32, half * 8, 16 + half * 8);
            acc[nb] = __builtin_amdgcn_wmma_f32_16x16x32_bf16(false, pf, false, vf, (short)0, acc[nb], false, false);
        }
    }

    // ---- epilogue: reduce row sums across lanes, normalize, store fp32 ----
    #pragma unroll
    for (int v = 0; v < 8; ++v) {
        float inv = 1.0f / red_sum16(rowsum[v]);
        int   m   = qrow + half * 8 + v;
        float* op = Ob + (size_t)m * DHEAD;
        #pragma unroll
        for (int nb = 0; nb < 8; ++nb)
            op[16 * nb + lq] = acc[nb][v] * inv;
    }
}

extern "C" void kernel_launch(void* const* d_in, const int* in_sizes, int n_in,
                              void* d_out, int out_size, void* d_ws, size_t ws_size,
                              hipStream_t stream) {
    // setup_inputs order: keys, queries, values, mask (mask is causal; handled analytically)
    const float* keys    = (const float*)d_in[0];
    const float* queries = (const float*)d_in[1];
    const float* values  = (const float*)d_in[2];
    float*       out     = (float*)d_out;

    const int BH = in_sizes[1] / (SLEN * DHEAD);  // B*H = 64
    dim3 grid(SLEN / 128, BH);
    fa_fwd_kernel<<<grid, 256, 0, stream>>>(keys, queries, values, out);
}
